// KANLinear_68607807586406
// MI455X (gfx1250) — matrix operand order repaired and checked by reading, not compiled, not run on hardware
//
#include <hip/hip_runtime.h>
#include <hip/hip_bf16.h>

// ---------------------------------------------------------------------------
// KANLinear forward for MI455X (gfx1250, wave32, WMMA).
//   y = einsum('bij,ioj->bo', B, coef*scale_sp) + einsum('bi,io->bo', silu(x), scale_base)
// Restructured as one GEMM:  Y[4096,512] = A[4096,6144](bf16) x Wt^T, with
//   A[b, i*12+j]  = Bspline_j(x[b,i]) (j<11),  A[b, i*12+11]  = silu(x[b,i])
//   Wt[o, i*12+j] = coef[i,o,j]*scale_sp[i,o], Wt[o, i*12+11] = scale_base[i,o]
// Wt is stored K-contiguous so both GEMM tiles stage into LDS as straight
// row-major b128 copies -> global_load_async_to_lds_b128 double-buffering.
// ---------------------------------------------------------------------------

#define BATCH 4096
#define IN_DIM 512
#define OUT_DIM 512
#define NSPL 11          // num + k = 8 + 3 basis functions
#define KPACK 12         // 11 spline + 1 base per input dim
#define KDIM (IN_DIM * KPACK)   // 6144
#define NGRID 15         // num + 1 + 2k grid points per input dim

typedef __attribute__((ext_vector_type(16))) __bf16 v16bf;
typedef __attribute__((ext_vector_type(8)))  float  v8f;

union FragAB {
    uint4 u4[2];
    v16bf v;
};

static __device__ __forceinline__ unsigned short f32_to_bf16_rne(float f) {
    unsigned int u = __float_as_uint(f);
    unsigned int lsb = (u >> 16) & 1u;
    u += 0x7FFFu + lsb;            // round-to-nearest-even
    return (unsigned short)(u >> 16);
}

// LDS byte offset of a __shared__ object (generic -> addrspace(3) -> int)
static __device__ __forceinline__ unsigned lds_addr_of(const void* p) {
    return (unsigned)(unsigned long long)(__attribute__((address_space(3))) const void*)p;
}

// gfx1250 async global->LDS copy, GVS addressing: SGPR base + 32-bit VGPR off.
static __device__ __forceinline__ void async_copy_b128(unsigned lds_byte_addr,
                                                       unsigned glb_byte_off,
                                                       const void* base) {
    asm volatile("global_load_async_to_lds_b128 %0, %1, %2 offset:0"
                 :: "v"(lds_byte_addr), "v"(glb_byte_off), "s"(base)
                 : "memory");
}

// ---------------------------------------------------------------------------
// Phase 1: per-(b,i) cubic B-spline basis (Cox-de Boor) + SiLU, pack to bf16.
// ---------------------------------------------------------------------------
__global__ __launch_bounds__(256) void spline_pack(
    const float* __restrict__ x,
    const float* __restrict__ grid,
    unsigned short* __restrict__ A)
{
    int idx = blockIdx.x * blockDim.x + threadIdx.x;
    if (idx >= BATCH * IN_DIM) return;
    int b = idx / IN_DIM;
    int i = idx - b * IN_DIM;

    float xv = x[(size_t)b * IN_DIM + i];

    float g[NGRID];
    #pragma unroll
    for (int t = 0; t < NGRID; ++t) g[t] = grid[i * NGRID + t];

    // degree-0 indicators: 14 intervals
    float B[NGRID - 1];
    #pragma unroll
    for (int t = 0; t < NGRID - 1; ++t)
        B[t] = (xv >= g[t] && xv < g[t + 1]) ? 1.0f : 0.0f;

    // Cox-de Boor recursion up to degree 3 (k = 3)
    for (int p = 1; p <= 3; ++p) {
        for (int t = 0; t < NGRID - 1 - p; ++t) {
            float d1 = g[t + p] - g[t];
            float d2 = g[t + p + 1] - g[t + 1];
            float t1 = (d1 != 0.0f) ? (xv - g[t]) / d1 * B[t] : 0.0f;
            float t2 = (d2 != 0.0f) ? (g[t + p + 1] - xv) / d2 * B[t + 1] : 0.0f;
            float r = t1 + t2;
            B[t] = (r == r) ? r : 0.0f;   // nan_to_num
        }
    }

    unsigned short* dst = A + (size_t)b * KDIM + (size_t)i * KPACK;
    #pragma unroll
    for (int j = 0; j < NSPL; ++j) dst[j] = f32_to_bf16_rne(B[j]);
    float silu = xv / (1.0f + __expf(-xv));
    dst[NSPL] = f32_to_bf16_rne(silu);
}

// ---------------------------------------------------------------------------
// Phase 2: fold scale_sp into coef, append scale_base; pack bf16 TRANSPOSED:
//          Wt[o, r] with r = i*12+j, K-contiguous rows of length 6144.
// ---------------------------------------------------------------------------
__global__ __launch_bounds__(256) void weight_pack(
    const float* __restrict__ coef,
    const float* __restrict__ scale_base,
    const float* __restrict__ scale_sp,
    unsigned short* __restrict__ Wt)
{
    int idx = blockIdx.x * blockDim.x + threadIdx.x;
    if (idx >= KDIM * OUT_DIM) return;
    int o = idx / KDIM;
    int r = idx - o * KDIM;       // r = i*12 + j (fastest -> contiguous writes)
    int i = r / KPACK;
    int j = r - i * KPACK;
    float v;
    if (j < NSPL)
        v = coef[((size_t)i * OUT_DIM + o) * NSPL + j] * scale_sp[(size_t)i * OUT_DIM + o];
    else
        v = scale_base[(size_t)i * OUT_DIM + o];
    Wt[(size_t)o * KDIM + r] = f32_to_bf16_rne(v);
}

// ---------------------------------------------------------------------------
// Phase 3: bf16 WMMA GEMM  C[M,N] = A[M,K] x Wt[N,K]^T,  f32 accumulate.
// 128x128 workgroup tile, BK=32, 8 waves (wave32) each own a 32x64 patch.
// Both tiles staged K-contiguous in LDS via async global->LDS b128 copies,
// double-buffered against the WMMA pipeline (ASYNCcnt ordering).
// ---------------------------------------------------------------------------
#define TM 128
#define TN 128
#define TK 32

__global__ __launch_bounds__(256) void kan_gemm(
    const unsigned short* __restrict__ A,
    const unsigned short* __restrict__ Wt,
    float* __restrict__ C)
{
    __shared__ __align__(16) unsigned short As[2][TM * TK];   // [row][k]
    __shared__ __align__(16) unsigned short Bs[2][TN * TK];   // [col][k]

    const int tid  = threadIdx.x;
    const int lane = tid & 31;
    const int wave = tid >> 5;      // 0..7
    const int wm   = wave >> 1;     // 0..3  -> 32-row slab
    const int wn   = wave & 1;      // 0..1  -> 64-col slab
    const int m0   = blockIdx.y * TM;
    const int n0   = blockIdx.x * TN;

    const int lr = lane & 15;       // 0..15 : M (A frag) / N (B frag) within tile
    const int lh = lane >> 4;       // lane half

    v8f acc[2][4];
    #pragma unroll
    for (int mt = 0; mt < 2; ++mt)
        #pragma unroll
        for (int nt = 0; nt < 4; ++nt)
            acc[mt][nt] = v8f{};

    // staging coordinates: each thread copies 16B at (row, col) and (row+64, col)
    const int sr = tid >> 2;              // 0..63
    const int sc = (tid & 3) * 8;         // 0,8,16,24 (ushorts)

    const unsigned asA = lds_addr_of(&As[0][0]);
    const unsigned asB = lds_addr_of(&Bs[0][0]);
    const unsigned ldsStride = (unsigned)(64 * TK * 2);     // 64 rows of tile
    const unsigned ldsThread = (unsigned)((sr * TK + sc) * 2);

    auto issue_tile = [&](int buf, int kkv) {
        const unsigned la = asA + (unsigned)buf * (TM * TK * 2) + ldsThread;
        const unsigned lb = asB + (unsigned)buf * (TN * TK * 2) + ldsThread;
        const unsigned ga = (unsigned)((((size_t)(m0 + sr)) * KDIM + kkv + sc) * 2);
        const unsigned gb = (unsigned)((((size_t)(n0 + sr)) * KDIM + kkv + sc) * 2);
        async_copy_b128(la,             ga,                        A);
        async_copy_b128(la + ldsStride, ga + 64u * KDIM * 2u,      A);
        async_copy_b128(lb,             gb,                        Wt);
        async_copy_b128(lb + ldsStride, gb + 64u * KDIM * 2u,      Wt);
    };

    issue_tile(0, 0);

    for (int kk = 0, it = 0; kk < KDIM; kk += TK, ++it) {
        const int buf = it & 1;
        if (kk + TK < KDIM) {
            issue_tile(buf ^ 1, kk + TK);                  // prefetch next tile
            if (kk + 2 * TK < KDIM)                        // L2 prefetch of K+2 slab
                __builtin_prefetch(A + ((size_t)(m0 + sr)) * KDIM + kk + 2 * TK + sc, 0, 1);
            // current buffer = 4 oldest async ops; async loads complete in order
            asm volatile("s_wait_asynccnt 0x4" ::: "memory");
        } else {
            asm volatile("s_wait_asynccnt 0x0" ::: "memory");
        }
        __syncthreads();

        // ---- fragment fill ----
        // A 16x32 bf16 layout: lane(M=lr, half=lh): elems 0..7 -> K=8*lh+e,
        //                      elems 8..15 -> K=16+8*lh+(e-8)  -> two b128 loads
        FragAB afrag[2];
        #pragma unroll
        for (int mt = 0; mt < 2; ++mt) {
            const int r = wm * 32 + mt * 16 + lr;
            afrag[mt].u4[0] = *(const uint4*)&As[buf][r * TK + 8 * lh];
            afrag[mt].u4[1] = *(const uint4*)&As[buf][r * TK + 16 + 8 * lh];
        }
        // B 32x16 bf16 layout: lane(N=lr, half=lh): elems 0..7 -> K=8*lh+e, elems 8..15 -> K=16+8*lh+(e-8)
        FragAB bfrag[4];
        #pragma unroll
        for (int nt = 0; nt < 4; ++nt) {
            const int c = wn * 64 + nt * 16 + lr;
            bfrag[nt].u4[0] = *(const uint4*)&Bs[buf][c * TK + 8 * lh];
            bfrag[nt].u4[1] = *(const uint4*)&Bs[buf][c * TK + 16 + 8 * lh];
        }

        // ---- 8 WMMAs per wave per K-step ----
        #pragma unroll
        for (int mt = 0; mt < 2; ++mt)
            #pragma unroll
            for (int nt = 0; nt < 4; ++nt)
                acc[mt][nt] = __builtin_amdgcn_wmma_f32_16x16x32_bf16(
                    false, afrag[mt].v, false, bfrag[nt].v,
                    (short)0, acc[mt][nt], false, false);

        __syncthreads();   // all waves done reading buf before it is rewritten
    }

    // ---- epilogue: C/D layout: VGPR v -> M = v + 8*lh, N = lr ----
    #pragma unroll
    for (int mt = 0; mt < 2; ++mt)
        #pragma unroll
        for (int nt = 0; nt < 4; ++nt)
            #pragma unroll
            for (int v = 0; v < 8; ++v) {
                const int row = m0 + wm * 32 + mt * 16 + v + 8 * lh;
                const int col = n0 + wn * 64 + nt * 16 + lr;
                C[(size_t)row * OUT_DIM + col] = acc[mt][nt][v];
            }
}

// ---------------------------------------------------------------------------
extern "C" void kernel_launch(void* const* d_in, const int* in_sizes, int n_in,
                              void* d_out, int out_size, void* d_ws, size_t ws_size,
                              hipStream_t stream) {
    const float* x          = (const float*)d_in[0];
    const float* grid       = (const float*)d_in[1];
    const float* coef       = (const float*)d_in[2];
    const float* scale_base = (const float*)d_in[3];
    const float* scale_sp   = (const float*)d_in[4];
    // d_in[5] = k (always 3, baked into the kernels)

    float* out = (float*)d_out;

    unsigned short* Abf = (unsigned short*)d_ws;                    // 4096*6144 bf16 = 48 MB
    unsigned short* Wtbf = Abf + (size_t)BATCH * KDIM;              // 512*6144  bf16 =  6 MB

    spline_pack<<<(BATCH * IN_DIM + 255) / 256, 256, 0, stream>>>(x, grid, Abf);
    weight_pack<<<(KDIM * OUT_DIM + 255) / 256, 256, 0, stream>>>(coef, scale_base, scale_sp, Wtbf);
    kan_gemm<<<dim3(OUT_DIM / TN, BATCH / TM), 256, 0, stream>>>(Abf, Wtbf, out);
}
